// DynamicsTransformer_6141803233372
// MI455X (gfx1250) — matrix operand order repaired and checked by reading, not compiled, not run on hardware
//
#include <hip/hip_runtime.h>
#include <cstdint>
#include <cstddef>

// ---------------------------------------------------------------------------
// Problem constants (match reference)
// ---------------------------------------------------------------------------
#define BB    2
#define TT    128
#define SSZ   256
#define DD    1024
#define HH    16
#define HKVN  4
#define HDIM  64
#define GQA   (HH / HKVN)     // 4
#define FFND  4096
#define CAPV  50.0f
#define EPSV  1e-6f
#define SCALEV 0.125f         // 64^-0.5

typedef __attribute__((ext_vector_type(16))) _Float16 v16h;
typedef __attribute__((ext_vector_type(8)))  float    v8f;

// ---- optional gfx1250 async global->LDS path (probe via __has_builtin) ----
#if defined(__has_builtin)
#if __has_builtin(__builtin_amdgcn_global_load_async_to_lds_b128)
#define HAVE_ASYNC_LDS 1
#endif
#endif
#ifndef HAVE_ASYNC_LDS
#define HAVE_ASYNC_LDS 0
#endif

#if HAVE_ASYNC_LDS
typedef int nat_int4 __attribute__((vector_size(16)));           // native int4
typedef __attribute__((address_space(1))) nat_int4 gas_int4;     // global int4
typedef __attribute__((address_space(3))) nat_int4 las_int4;     // LDS int4
#endif

__device__ __forceinline__ void async_wait_all()
{
#if HAVE_ASYNC_LDS
#if __has_builtin(__builtin_amdgcn_s_wait_asynccnt)
  __builtin_amdgcn_s_wait_asynccnt(0);
#else
  asm volatile("s_wait_asynccnt 0x0" ::: "memory");
#endif
#endif
}

// ---------------------------------------------------------------------------
// Tiled WMMA GEMM:  C[M,N] = residual + f16(A') @ W[N,K]^T
//   MODE 0: A' = rowscale[row] * colscale[k] * A32[M,K]   (fused rmsnorm)
//   MODE 1: A' = A32[M,K]
//   MODE 2: A' = A16[M,K]  (already f16)
// Requirements (hold for every call site): M%128==0, N%64==0, K%32==0.
// BM=128, BN=64, BK=32, 256 threads = 8 waves; each wave owns a 32x32 patch
// via 4x v_wmma_f32_16x16x32_f16 per K step. Double-buffered LDS; B tile
// (and A tile in MODE 2) staged with async global->LDS when available.
// ---------------------------------------------------------------------------
#define BM 128
#define BN 64
#define BK 32
#define LDA 40   // halves per A row in LDS (padded)
#define LDB 40   // halves per B row in LDS (padded)

template <int MODE>
__global__ __launch_bounds__(256)
void gemm_wmma_kernel(const float* __restrict__ A32,
                      const _Float16* __restrict__ A16,
                      const _Float16* __restrict__ W,
                      const float* __restrict__ rowscale,
                      const float* __restrict__ colscale,
                      const float* __restrict__ residual,
                      float* __restrict__ C32,
                      _Float16* __restrict__ C16,
                      int M, int N, int K)
{
  __shared__ _Float16 sA[2][BM * LDA];   // 2 x 10240 B
  __shared__ _Float16 sB[2][BN * LDB];   // 2 x  5120 B

  const int tid  = threadIdx.x;
  const int lane = tid & 31;
  const int wave = tid >> 5;
  const int wm   = wave & 3;        // 0..3 -> 32-row group
  const int wn   = wave >> 2;       // 0..1 -> 32-col group
  const int bm   = blockIdx.y * BM;
  const int bn   = blockIdx.x * BN;

  // A tile loader: 2 threads per row, 16 halves each
  const int ar  = tid >> 1;         // 0..127
  const int ac0 = (tid & 1) << 4;   // 0 or 16
  // B tile loader: 4 threads per row, 8 halves each
  const int br  = tid >> 2;         // 0..63
  const int bc0 = (tid & 3) << 3;   // 0,8,16,24

  const size_t aRow = (size_t)(bm + ar) * K;
  const size_t bRow = (size_t)(bn + br) * K;

  float rs = 1.0f;
  if constexpr (MODE == 0) rs = rowscale[bm + ar];

  v8f acc[2][2] = {};
  union Frag { v16h v; uint4 q[2]; };

  // prefetch registers (register path only)
  float4 pa[4];
  float4 pc[4];
  uint4  pa16[2];
  uint4  pb;

  auto fetch = [&](int step, int buf) {
    const int kk = step * BK;
#if HAVE_ASYNC_LDS
    (void)buf;
    __builtin_amdgcn_global_load_async_to_lds_b128(
        (gas_int4*)(W + bRow + kk + bc0),
        (las_int4*)&sB[buf][br * LDB + bc0], 0, 0);
    if constexpr (MODE == 2) {
      const _Float16* p = A16 + aRow + kk + ac0;
      _Float16* d = &sA[buf][ar * LDA + ac0];
      __builtin_amdgcn_global_load_async_to_lds_b128(
          (gas_int4*)p, (las_int4*)d, 0, 0);
      __builtin_amdgcn_global_load_async_to_lds_b128(
          (gas_int4*)(p + 8), (las_int4*)(d + 8), 0, 0);
    }
#else
    (void)buf;
    pb = *(const uint4*)(W + bRow + kk + bc0);
    if constexpr (MODE == 2) {
      const _Float16* p = A16 + aRow + kk + ac0;
      pa16[0] = ((const uint4*)p)[0];
      pa16[1] = ((const uint4*)p)[1];
    }
#endif
    if constexpr (MODE != 2) {
      const float* p = A32 + aRow + kk + ac0;
      pa[0] = ((const float4*)p)[0];
      pa[1] = ((const float4*)p)[1];
      pa[2] = ((const float4*)p)[2];
      pa[3] = ((const float4*)p)[3];
      if constexpr (MODE == 0) {
        const float* c = colscale + kk + ac0;
        pc[0] = ((const float4*)c)[0];
        pc[1] = ((const float4*)c)[1];
        pc[2] = ((const float4*)c)[2];
        pc[3] = ((const float4*)c)[3];
      }
    }
  };

  auto stage = [&](int buf) {
    uint4* dstA = (uint4*)&sA[buf][ar * LDA + ac0];
    if constexpr (MODE == 2) {
#if !HAVE_ASYNC_LDS
      dstA[0] = pa16[0];
      dstA[1] = pa16[1];
#endif
      (void)dstA;
    } else {
      union { _Float16 h[16]; uint4 q[2]; } t;
      const float* af = (const float*)pa;
      const float* cf = (const float*)pc;
#pragma unroll
      for (int i = 0; i < 16; ++i) {
        float v = af[i];
        if constexpr (MODE == 0) v *= rs * cf[i];
        t.h[i] = (_Float16)v;
      }
      dstA[0] = t.q[0];
      dstA[1] = t.q[1];
    }
#if !HAVE_ASYNC_LDS
    *(uint4*)&sB[buf][br * LDB + bc0] = pb;
#endif
  };

  const int nsteps = K / BK;
  fetch(0, 0);
  stage(0);
  async_wait_all();
  __syncthreads();

  const int hb = (lane >> 4) << 3;     // 0 or 8
  for (int step = 0; step < nsteps; ++step) {
    const int cur = step & 1;
    const bool more = (step + 1) < nsteps;
    if (more) fetch(step + 1, (step + 1) & 1);

    // A fragment (16x32 f16): kA = (v/4)*16 + (lane/16)*8 + (v%4)*2 -> 2 b128
    // B fragment (32x16 f16): kB = (v/4)*16 + (lane/16)*8 + (v%4)*2 -> 2 b128
    Frag af[2], bf[2];
#pragma unroll
    for (int mi = 0; mi < 2; ++mi) {
      const _Float16* p = &sA[cur][(wm * 32 + mi * 16 + (lane & 15)) * LDA + hb];
      af[mi].q[0] = *(const uint4*)p;
      af[mi].q[1] = *(const uint4*)(p + 16);
    }
#pragma unroll
    for (int ni = 0; ni < 2; ++ni) {
      const _Float16* p = &sB[cur][(wn * 32 + ni * 16 + (lane & 15)) * LDB + hb];
      bf[ni].q[0] = *(const uint4*)p;
      bf[ni].q[1] = *(const uint4*)(p + 16);
    }
#pragma unroll
    for (int mi = 0; mi < 2; ++mi)
#pragma unroll
      for (int ni = 0; ni < 2; ++ni)
        acc[mi][ni] = __builtin_amdgcn_wmma_f32_16x16x32_f16(
            false, af[mi].v, false, bf[ni].v, (short)0, acc[mi][ni], false, false);

    if (more) stage((step + 1) & 1);
    if (more) async_wait_all();
    __syncthreads();
  }

  // epilogue: C layout lanes 0-15 rows 0..7 (vgpr r), lanes 16-31 rows 8..15
#pragma unroll
  for (int mi = 0; mi < 2; ++mi) {
#pragma unroll
    for (int ni = 0; ni < 2; ++ni) {
#pragma unroll
      for (int r = 0; r < 8; ++r) {
        const int row = bm + wm * 32 + mi * 16 + ((lane >> 4) << 3) + r;
        const int col = bn + wn * 32 + ni * 16 + (lane & 15);
        float v = acc[mi][ni][r];
        if (residual != nullptr) v += residual[(size_t)row * N + col];
        if (C32 != nullptr) C32[(size_t)row * N + col] = v;
        else                C16[(size_t)row * N + col] = (_Float16)v;
      }
    }
  }
}

// ---------------------------------------------------------------------------
// Per-row 1/rms over D=1024: one block of 256 per row, one float4 per lane.
// ---------------------------------------------------------------------------
__global__ __launch_bounds__(256)
void invrms_kernel(const float* __restrict__ X, float* __restrict__ out)
{
  __shared__ float red[256];
  const float4 v = ((const float4*)(X + (size_t)blockIdx.x * DD))[threadIdx.x];
  red[threadIdx.x] = v.x * v.x + v.y * v.y + v.z * v.z + v.w * v.w;
  __syncthreads();
  for (int o = 128; o > 0; o >>= 1) {
    if (threadIdx.x < o) red[threadIdx.x] += red[threadIdx.x + o];
    __syncthreads();
  }
  if (threadIdx.x == 0) out[blockIdx.x] = rsqrtf(red[0] / (float)DD + EPSV);
}

// ---------------------------------------------------------------------------
// f32 -> f16 copy, 4 elements per thread (n % 4 == 0 for all weights)
// ---------------------------------------------------------------------------
__global__ __launch_bounds__(256)
void f32_to_f16_kernel(const float* __restrict__ src, _Float16* __restrict__ dst, size_t n4)
{
  const size_t i = (size_t)blockIdx.x * 256 + threadIdx.x;
  if (i >= n4) return;
  const float4 v = ((const float4*)src)[i];
  union { _Float16 h[4]; uint2 u; } t;
  t.h[0] = (_Float16)v.x; t.h[1] = (_Float16)v.y;
  t.h[2] = (_Float16)v.z; t.h[3] = (_Float16)v.w;
  ((uint2*)dst)[i] = t.u;
}

// ---------------------------------------------------------------------------
// Cross attention: one block per (b,t,h). 256 threads (one per key s).
// Q: (B*T, 1024) f32 ; KV: (B*T*S, 512) f16, K cols [0,256), V cols [256,512)
// ---------------------------------------------------------------------------
__global__ __launch_bounds__(256)
void cross_attn_kernel(const float* __restrict__ Q,
                       const _Float16* __restrict__ KV,
                       const float* __restrict__ qn_w,
                       const float* __restrict__ kn_w,
                       float* __restrict__ out)
{
  const int bt  = blockIdx.x / HH;
  const int h   = blockIdx.x % HH;
  const int kvh = h / GQA;
  const int tid = threadIdx.x;

  __shared__ float qkn[HDIM];
  __shared__ float red[256];
  __shared__ float p[SSZ];
  __shared__ float part[4][HDIM];

  // q rmsnorm; fold attention scale and kn_w into the q vector
  const float* q = Q + (size_t)bt * (HH * HDIM) + h * HDIM;
  float ss = 0.0f;
  if (tid < HDIM) { float v = q[tid]; ss = v * v; }
  red[tid] = ss;
  __syncthreads();
  for (int o = 128; o > 0; o >>= 1) {
    if (tid < o) red[tid] += red[tid + o];
    __syncthreads();
  }
  const float qscale = rsqrtf(red[0] / (float)HDIM + EPSV) * SCALEV;
  __syncthreads();
  if (tid < HDIM) qkn[tid] = q[tid] * qscale * qn_w[tid] * kn_w[tid];
  __syncthreads();

  // score for key s = tid: packed b128 reads of the f16 K row
  const _Float16* krow = KV + ((size_t)bt * SSZ + tid) * 512 + kvh * HDIM;
  union U8 { uint4 q; _Float16 h[8]; };
  float ksum = 0.0f, dot = 0.0f;
#pragma unroll
  for (int c = 0; c < 8; ++c) {
    U8 u; u.q = ((const uint4*)krow)[c];
#pragma unroll
    for (int j = 0; j < 8; ++j) {
      const float kv = (float)u.h[j];
      ksum += kv * kv;
      dot  += qkn[c * 8 + j] * kv;
    }
  }
  float sc = dot * rsqrtf(ksum / (float)HDIM + EPSV);
  sc = CAPV * tanhf(sc / CAPV);

  // softmax over 256 keys
  red[tid] = sc;
  __syncthreads();
  for (int o = 128; o > 0; o >>= 1) {
    if (tid < o) red[tid] = fmaxf(red[tid], red[tid + o]);
    __syncthreads();
  }
  const float mx = red[0];
  __syncthreads();
  const float e = __expf(sc - mx);
  red[tid] = e;
  __syncthreads();
  for (int o = 128; o > 0; o >>= 1) {
    if (tid < o) red[tid] += red[tid + o];
    __syncthreads();
  }
  const float inv = 1.0f / red[0];
  __syncthreads();
  p[tid] = e * inv;
  __syncthreads();

  // out[hd] = sum_s p[s] * V[s][hd] ; 4 groups of 64 keys, then reduce
  const int hd  = tid & 63;
  const int grp = tid >> 6;
  float acc = 0.0f;
  for (int s2 = grp * 64; s2 < grp * 64 + 64; ++s2) {
    const _Float16* vrow = KV + ((size_t)bt * SSZ + s2) * 512 + 256 + kvh * HDIM;
    acc += p[s2] * (float)vrow[hd];
  }
  part[grp][hd] = acc;
  __syncthreads();
  if (tid < HDIM)
    out[(size_t)bt * (HH * HDIM) + h * HDIM + tid] =
        part[0][tid] + part[1][tid] + part[2][tid] + part[3][tid];
}

// ---------------------------------------------------------------------------
// Temporal (causal) attention: one block per (b,h), 128 threads (one per query).
// Qp: (B*T, 1024) f32 ; KVp: (B*T, 512) f32.
// ---------------------------------------------------------------------------
__global__ __launch_bounds__(128)
void temporal_attn_kernel(const float* __restrict__ Qp,
                          const float* __restrict__ KVp,
                          const float* __restrict__ qn_w,
                          const float* __restrict__ kn_w,
                          float* __restrict__ out)
{
  const int b   = blockIdx.x / HH;
  const int h   = blockIdx.x % HH;
  const int kvh = h / GQA;
  const int t   = threadIdx.x;  // 0..127

  __shared__ _Float16 Kn[TT][HDIM];  // 16 KB
  __shared__ _Float16 Vl[TT][HDIM];  // 16 KB

  // stage normalized K row t and V row t
  {
    const float* kr = KVp + (size_t)(b * TT + t) * 512 + kvh * HDIM;
    const float* vr = kr + 256;
    float ks = 0.0f;
#pragma unroll 8
    for (int j = 0; j < HDIM; ++j) ks += kr[j] * kr[j];
    const float kscale = rsqrtf(ks / (float)HDIM + EPSV);
#pragma unroll 8
    for (int j = 0; j < HDIM; ++j) {
      Kn[t][j] = (_Float16)(kr[j] * kscale * kn_w[j]);
      Vl[t][j] = (_Float16)vr[j];
    }
  }
  __syncthreads();

  // q row (rmsnorm, fold scale)
  float qv[HDIM];
  {
    const float* qr = Qp + (size_t)(b * TT + t) * (HH * HDIM) + h * HDIM;
    float qs = 0.0f;
#pragma unroll 8
    for (int j = 0; j < HDIM; ++j) { qv[j] = qr[j]; qs += qv[j] * qv[j]; }
    const float qscale = rsqrtf(qs / (float)HDIM + EPSV) * SCALEV;
#pragma unroll 8
    for (int j = 0; j < HDIM; ++j) qv[j] *= qscale * qn_w[j];
  }

  // pass 1: row max of soft-capped scores over k <= t
  float mx = -3.0e38f;
  for (int k = 0; k <= t; ++k) {
    float d = 0.0f;
#pragma unroll 8
    for (int j = 0; j < HDIM; ++j) d += qv[j] * (float)Kn[k][j];
    d = CAPV * tanhf(d / CAPV);
    mx = fmaxf(mx, d);
  }
  // pass 2: exp and accumulate
  float o[HDIM];
#pragma unroll 8
  for (int j = 0; j < HDIM; ++j) o[j] = 0.0f;
  float sum = 0.0f;
  for (int k = 0; k <= t; ++k) {
    float d = 0.0f;
#pragma unroll 8
    for (int j = 0; j < HDIM; ++j) d += qv[j] * (float)Kn[k][j];
    d = CAPV * tanhf(d / CAPV);
    const float e = __expf(d - mx);
    sum += e;
#pragma unroll 8
    for (int j = 0; j < HDIM; ++j) o[j] += e * (float)Vl[k][j];
  }
  const float inv = 1.0f / sum;
#pragma unroll 8
  for (int j = 0; j < HDIM; ++j)
    out[(size_t)(b * TT + t) * (HH * HDIM) + h * HDIM + j] = o[j] * inv;
}

// ---------------------------------------------------------------------------
// SwiGLU gate: Hf[r,c] = f16( silu(GU[r,c]) * GU[r, FFN+c] ), 4 elems / thread
// ---------------------------------------------------------------------------
__global__ __launch_bounds__(256)
void swiglu_kernel(const float* __restrict__ GU, _Float16* __restrict__ Hf, size_t n4)
{
  const size_t i = (size_t)blockIdx.x * 256 + threadIdx.x;
  if (i >= n4) return;
  const size_t e0 = i * 4;
  const size_t r  = e0 >> 12;           // / 4096
  const size_t c  = e0 & 4095;
  const float4 g = *(const float4*)(GU + r * (2 * FFND) + c);
  const float4 u = *(const float4*)(GU + r * (2 * FFND) + FFND + c);
  union { _Float16 h[4]; uint2 q; } t;
  t.h[0] = (_Float16)((g.x / (1.0f + __expf(-g.x))) * u.x);
  t.h[1] = (_Float16)((g.y / (1.0f + __expf(-g.y))) * u.y);
  t.h[2] = (_Float16)((g.z / (1.0f + __expf(-g.z))) * u.z);
  t.h[3] = (_Float16)((g.w / (1.0f + __expf(-g.w))) * u.w);
  ((uint2*)Hf)[i] = t.q;
}

// ---------------------------------------------------------------------------
// Host-side orchestration
// ---------------------------------------------------------------------------
extern "C" void kernel_launch(void* const* d_in, const int* in_sizes, int n_in,
                              void* d_out, int out_size, void* d_ws, size_t ws_size,
                              hipStream_t stream)
{
  (void)in_sizes; (void)n_in; (void)out_size; (void)ws_size;

  const float* agent    = (const float*)d_in[0];
  const float* z        = (const float*)d_in[1];
  const float* norm1_w  = (const float*)d_in[2];
  const float* norm_kv_w= (const float*)d_in[3];
  const float* cq_w     = (const float*)d_in[4];
  const float* ck_w     = (const float*)d_in[5];
  const float* cv_w     = (const float*)d_in[6];
  const float* co_w     = (const float*)d_in[7];
  const float* c_qn_w   = (const float*)d_in[8];
  const float* c_kn_w   = (const float*)d_in[9];
  const float* norm2_w  = (const float*)d_in[10];
  const float* sq_w     = (const float*)d_in[11];
  const float* sk_w     = (const float*)d_in[12];
  const float* sv_w     = (const float*)d_in[13];
  const float* so_w     = (const float*)d_in[14];
  const float* s_qn_w   = (const float*)d_in[15];
  const float* s_kn_w   = (const float*)d_in[16];
  const float* norm3_w  = (const float*)d_in[17];
  const float* w1       = (const float*)d_in[18];
  const float* w3       = (const float*)d_in[19];
  const float* w2w      = (const float*)d_in[20];

  char* base = (char*)d_ws;
  size_t off = 0;
  auto alloc = [&](size_t bytes) -> void* {
    off = (off + 255) & ~(size_t)255;
    void* p = base + off;
    off += bytes;
    return p;
  };

  const size_t BT  = (size_t)BB * TT;          // 256
  const size_t BTS = BT * SSZ;                 // 65536

  _Float16* cq16   = (_Float16*)alloc((size_t)1024 * 1024 * 2);
  _Float16* ckcv16 = (_Float16*)alloc((size_t)512  * 1024 * 2);
  _Float16* co16   = (_Float16*)alloc((size_t)1024 * 1024 * 2);
  _Float16* sq16   = (_Float16*)alloc((size_t)1024 * 1024 * 2);
  _Float16* skv16  = (_Float16*)alloc((size_t)512  * 1024 * 2);
  _Float16* so16   = (_Float16*)alloc((size_t)1024 * 1024 * 2);
  _Float16* w13_16 = (_Float16*)alloc((size_t)8192 * 1024 * 2);
  _Float16* w2_16  = (_Float16*)alloc((size_t)1024 * 4096 * 2);
  float*    r1   = (float*)alloc(BT * 4);
  float*    rz   = (float*)alloc(BTS * 4);
  float*    r2   = (float*)alloc(BT * 4);
  float*    r3   = (float*)alloc(BT * 4);
  float*    qbuf = (float*)alloc(BT * 1024 * 4);
  _Float16* kv16 = (_Float16*)alloc(BTS * 512 * 2);
  float*    atto = (float*)alloc(BT * 1024 * 4);
  float*    x1   = (float*)alloc(BT * 1024 * 4);
  float*    sqb  = (float*)alloc(BT * 1024 * 4);
  float*    skvb = (float*)alloc(BT * 512 * 4);
  float*    tout = (float*)alloc(BT * 1024 * 4);
  float*    x2   = (float*)alloc(BT * 1024 * 4);
  float*    gu   = (float*)alloc(BT * 8192 * 4);
  _Float16* h16  = (_Float16*)alloc(BT * 4096 * 2);

  auto cvt = [&](const float* s, _Float16* d, size_t n) {
    const size_t n4 = n / 4;
    f32_to_f16_kernel<<<dim3((unsigned)((n4 + 255) / 256)), 256, 0, stream>>>(s, d, n4);
  };
  auto gemm = [&](int mode, const float* A32, const _Float16* A16, const _Float16* W,
                  const float* rs, const float* cs, const float* res,
                  float* C32, _Float16* C16, int M, int N, int K) {
    dim3 g((unsigned)(N / BN), (unsigned)(M / BM));
    if (mode == 0)
      gemm_wmma_kernel<0><<<g, 256, 0, stream>>>(A32, A16, W, rs, cs, res, C32, C16, M, N, K);
    else if (mode == 1)
      gemm_wmma_kernel<1><<<g, 256, 0, stream>>>(A32, A16, W, rs, cs, res, C32, C16, M, N, K);
    else
      gemm_wmma_kernel<2><<<g, 256, 0, stream>>>(A32, A16, W, rs, cs, res, C32, C16, M, N, K);
  };

  // ---- weight conversion (K/V and w1/w3 concatenated on N) ----
  cvt(cq_w, cq16, (size_t)1024 * 1024);
  cvt(ck_w, ckcv16,                (size_t)256 * 1024);
  cvt(cv_w, ckcv16 + 256 * 1024,   (size_t)256 * 1024);
  cvt(co_w, co16, (size_t)1024 * 1024);
  cvt(sq_w, sq16, (size_t)1024 * 1024);
  cvt(sk_w, skv16,                 (size_t)256 * 1024);
  cvt(sv_w, skv16 + 256 * 1024,    (size_t)256 * 1024);
  cvt(so_w, so16, (size_t)1024 * 1024);
  cvt(w1,  w13_16,                       (size_t)4096 * 1024);
  cvt(w3,  w13_16 + (size_t)4096 * 1024, (size_t)4096 * 1024);
  cvt(w2w, w2_16, (size_t)1024 * 4096);

  // ---- layer 1: cross attention ----
  invrms_kernel<<<(unsigned)BT,  256, 0, stream>>>(agent, r1);
  invrms_kernel<<<(unsigned)BTS, 256, 0, stream>>>(z, rz);
  gemm(0, agent, nullptr, cq16, r1, norm1_w, nullptr, qbuf, nullptr, (int)BT, 1024, 1024);
  gemm(0, z, nullptr, ckcv16, rz, norm_kv_w, nullptr, nullptr, kv16, (int)BTS, 512, 1024);
  cross_attn_kernel<<<BB * TT * HH, 256, 0, stream>>>(qbuf, kv16, c_qn_w, c_kn_w, atto);
  gemm(1, atto, nullptr, co16, nullptr, nullptr, agent, x1, nullptr, (int)BT, 1024, 1024);

  // ---- layer 2: temporal attention ----
  invrms_kernel<<<(unsigned)BT, 256, 0, stream>>>(x1, r2);
  gemm(0, x1, nullptr, sq16,  r2, norm2_w, nullptr, sqb,  nullptr, (int)BT, 1024, 1024);
  gemm(0, x1, nullptr, skv16, r2, norm2_w, nullptr, skvb, nullptr, (int)BT, 512,  1024);
  temporal_attn_kernel<<<BB * HH, TT, 0, stream>>>(sqb, skvb, s_qn_w, s_kn_w, tout);
  gemm(1, tout, nullptr, so16, nullptr, nullptr, x1, x2, nullptr, (int)BT, 1024, 1024);

  // ---- FFN (SwiGLU) ----
  invrms_kernel<<<(unsigned)BT, 256, 0, stream>>>(x2, r3);
  gemm(0, x2, nullptr, w13_16, r3, norm3_w, nullptr, gu, nullptr, (int)BT, 8192, 1024);
  swiglu_kernel<<<(unsigned)((BT * 4096 / 4 + 255) / 256), 256, 0, stream>>>(gu, h16, BT * 4096 / 4);
  gemm(2, nullptr, h16, w2_16, nullptr, nullptr, x2, (float*)d_out, nullptr, (int)BT, 1024, 4096);
}
